// TransformerBlock_5970004541647
// MI455X (gfx1250) — matrix operand order repaired and checked by reading, not compiled, not run on hardware
//
#include <hip/hip_runtime.h>
#include <hip/hip_bf16.h>
#include <math.h>

typedef __bf16 bf16;
typedef bf16 v16bf __attribute__((ext_vector_type(16)));
typedef bf16 v8bf  __attribute__((ext_vector_type(8)));
typedef float v8f  __attribute__((ext_vector_type(8)));

#define NEG_INF (-1e30f)

// ---- WMMA fragment helpers (gfx1250, wave32) ----
// A-matrix 16x32 bf16 layout: lane L (m = L&15), elements i hold
// K = 16*(L>=16) + i (contiguous, matching ldfragB).
static __device__ inline v16bf ldfragA(const bf16* p) {
  v8bf a = *(const v8bf*)p;
  v8bf b = *(const v8bf*)(p + 8);
  return __builtin_shufflevector(a, b, 0,1,2,3,4,5,6,7,8,9,10,11,12,13,14,15);
}
// B-matrix 32x16 bf16: lane L (n = L&15), elements i hold K = 16*(L>=16) + i,
// i.e. 16 contiguous K values per lane when stored [n][k].
static __device__ inline v16bf ldfragB(const bf16* p) {
  v8bf a = *(const v8bf*)p;
  v8bf b = *(const v8bf*)(p + 8);
  return __builtin_shufflevector(a, b, 0,1,2,3,4,5,6,7,8,9,10,11,12,13,14,15);
}
static __device__ inline v8f wmma_bf16(v16bf a, v16bf b, v8f c) {
  return __builtin_amdgcn_wmma_f32_16x16x32_bf16(false, a, false, b, (short)0, c,
                                                 false, false);
}
static __device__ inline v8f vzero8() {
  v8f z = {0.f,0.f,0.f,0.f,0.f,0.f,0.f,0.f};
  return z;
}

// ---- CDNA5 async global->LDS copy (ASYNCcnt-tracked, §15.18.3 op 98) ----
// LDS aperture keeps the LDS byte offset in addr[31:0], so truncating the
// generic __shared__ pointer yields the per-wave LDS address the op wants.
static __device__ inline uint32_t lds_addr(const void* p) {
  return (uint32_t)(uintptr_t)p;
}
static __device__ inline void async_copy16(uint32_t lds, const bf16* g) {
  asm volatile("global_load_async_to_lds_b128 %0, %1, off"
               :: "v"(lds), "v"((uint64_t)(uintptr_t)g) : "memory");
}
static __device__ inline void wait_async0() {
  asm volatile("s_wait_asynccnt 0x0" ::: "memory");
}

// ---- fp32 -> bf16 transposed convert: d[n*K+k] = (bf16)s[k*N+n] ----
__global__ void cvtT_kernel(const float* __restrict__ s, bf16* __restrict__ d,
                            int K, int N) {
  int idx = blockIdx.x * 256 + threadIdx.x;
  if (idx < K * N) {
    int n = idx / K, k = idx - n * K;
    d[idx] = (bf16)s[(size_t)k * N + n];
  }
}

// ---- LayerNorm (one block per token), fp32 in -> bf16 out ----
__global__ __launch_bounds__(256) void ln_kernel(const float* __restrict__ x,
                                                 const float* __restrict__ w,
                                                 const float* __restrict__ b,
                                                 bf16* __restrict__ out, int C) {
  const int row = blockIdx.x;
  const int tid = threadIdx.x;
  __shared__ float red[256];
  const float* xr = x + (size_t)row * C;

  float s = 0.f;
  for (int c = tid; c < C; c += 256) s += xr[c];
  red[tid] = s; __syncthreads();
  for (int st = 128; st > 0; st >>= 1) { if (tid < st) red[tid] += red[tid + st]; __syncthreads(); }
  float mean = red[0] / (float)C;
  __syncthreads();

  float s2 = 0.f;
  for (int c = tid; c < C; c += 256) { float d0 = xr[c] - mean; s2 += d0 * d0; }
  red[tid] = s2; __syncthreads();
  for (int st = 128; st > 0; st >>= 1) { if (tid < st) red[tid] += red[tid + st]; __syncthreads(); }
  float rstd = rsqrtf(red[0] / (float)C + 1e-5f);

  for (int c = tid; c < C; c += 256)
    out[(size_t)row * C + c] = (bf16)((xr[c] - mean) * rstd * w[c] + b[c]);
}

// ---- bf16 WMMA GEMM: out = act(A[M,K] @ Bt[N,K]^T + bias) (+ resid) ----
// Block tile 64x64, BK=64, double-buffered async LDS staging overlapping the
// WMMA compute. 8 waves (4m x 2n), each wave: 16 rows x two 16x16 tiles.
__global__ __launch_bounds__(256) void gemm_kernel(
    const bf16* __restrict__ A,   // [M,K] row-major
    const bf16* __restrict__ Bt,  // [N,K] row-major (pre-transposed weights)
    const float* __restrict__ bias, const float* __restrict__ resid,
    float* __restrict__ outF, bf16* __restrict__ outB,
    int M, int N, int K, int act)
{
  __shared__ __align__(16) bf16 As[2][64][64];
  __shared__ __align__(16) bf16 Bs[2][64][64];
  const int tid  = threadIdx.x;
  const int lane = tid & 31;
  const int wave = tid >> 5;
  const int wm = wave >> 1, wn = wave & 1;
  const int m0 = blockIdx.y * 64, n0 = blockIdx.x * 64;
  const int hi = lane >> 4, lo = lane & 15;

  // staging map: thread t copies 32B (two b128) of one row of each matrix
  const int srow  = tid >> 2;          // 0..63
  const int scoff = (tid & 3) * 16;    // element offset 0/16/32/48

  v8f acc0 = vzero8();
  v8f acc1 = vzero8();

  auto stage = [&](int buf, int k0) {
    const bf16* ga = &A[(size_t)(m0 + srow) * K + k0 + scoff];
    uint32_t la = lds_addr(&As[buf][srow][scoff]);
    async_copy16(la,      ga);
    async_copy16(la + 16, ga + 8);
    const bf16* gb = &Bt[(size_t)(n0 + srow) * K + k0 + scoff];
    uint32_t lb = lds_addr(&Bs[buf][srow][scoff]);
    async_copy16(lb,      gb);
    async_copy16(lb + 16, gb + 8);
  };

  const int steps = K >> 6;
  stage(0, 0);
  for (int i = 0; i < steps; i++) {
    wait_async0();        // own async stores for stage i complete
    __syncthreads();      // everyone's stage i visible; prev reads of other buf done
    if (i + 1 < steps) stage((i + 1) & 1, (i + 1) << 6);  // overlaps compute below
    const int b = i & 1;
    #pragma unroll
    for (int kk = 0; kk < 2; kk++) {
      v16bf af  = ldfragA(&As[b][wm * 16 + lo][kk * 32 + hi * 16]);
      v16bf bf0 = ldfragB(&Bs[b][wn * 32 + lo][kk * 32 + hi * 16]);
      v16bf bf1 = ldfragB(&Bs[b][wn * 32 + 16 + lo][kk * 32 + hi * 16]);
      acc0 = wmma_bf16(af, bf0, acc0);
      acc1 = wmma_bf16(af, bf1, acc1);
    }
  }

  #pragma unroll
  for (int r = 0; r < 8; r++) {
    int gm = m0 + wm * 16 + r + hi * 8;
    #pragma unroll
    for (int nt = 0; nt < 2; nt++) {
      int gn = n0 + wn * 32 + nt * 16 + lo;
      float v = (nt == 0 ? acc0[r] : acc1[r]) + bias[gn];
      if (act) v = 0.5f * v * (1.0f + erff(v * 0.70710678118f));  // exact GELU
      if (resid) v += resid[(size_t)gm * N + gn];
      if (outF) outF[(size_t)gm * N + gn] = v;
      if (outB) outB[(size_t)gm * N + gn] = (bf16)v;
    }
  }
}

// ---- Flash attention with future band ----
// One wave per (head, 16-row query block). Online softmax over 2048 columns,
// streamed 32 keys/iteration. Band columns c in (t, t+16] take fattn scores
// (precomputed into LDS), contribute to max/sum but not to P@V; their output
// contribution against fv_w is applied at the end.
__global__ __launch_bounds__(32) void attn_kernel(
    const bf16* __restrict__ qkv,    // [T, 2304], q|k|v at col 0/768/1536 + h*64
    const float* __restrict__ fkw,   // [12, 64, 2047]
    const float* __restrict__ fkb,   // [12, 1, 2047]
    const float* __restrict__ fvw,   // [12, 2047, 64]
    const float* __restrict__ fvb,   // [12, 1, 64]
    bf16* __restrict__ out, int T)   // [T, 768]
{
  const int h  = blockIdx.x;
  const int t0 = blockIdx.y * 16;
  const int lane = threadIdx.x;
  const int hi = lane >> 4, lo = lane & 15;
  const int C3 = 2304, Cc = 768, HS = 64;
  const int F1 = T - 1;              // 2047
  const float scale = 0.125f;        // 64^-0.5

  __shared__ __align__(16) bf16  Qs[16][64];   // q block (async-staged)
  __shared__ __align__(16) float fq[16][32];   // band scores, col f-t0 in [0,30]
  __shared__ __align__(16) float Sl[16][32];
  __shared__ __align__(16) bf16  Pl[16][32];
  __shared__ __align__(16) bf16  Vt[64][32];   // V transposed [d][key]
  __shared__ __align__(16) float Ol[16][64];
  __shared__ float mrow[16], lrow[16], rsc[16];

  // async-stage the q block: 16 rows x 128B = 128 chunks, 4 per lane
  #pragma unroll
  for (int j = 0; j < 4; j++) {
    int c = lane * 4 + j;
    int row = c >> 3, off = (c & 7) * 8;
    async_copy16(lds_addr(&Qs[row][off]),
                 &qkv[(size_t)(t0 + row) * C3 + h * HS + off]);
  }
  if (lane < 16) { mrow[lane] = NEG_INF; lrow[lane] = 0.f; }
  wait_async0();
  __syncthreads();

  // q A-fragments (K = 0..31 and 32..63) from LDS
  v16bf qa0 = ldfragA(&Qs[lo][hi * 16]);
  v16bf qa1 = ldfragA(&Qs[lo][32 + hi * 16]);

  // precompute future-band scores fq[r][f-t0], f in [t0, t0+30]
  {
    int f = t0 + lane;
    for (int r = 0; r < 16; r++) {
      float val = NEG_INF;
      if (lane < 31 && f < F1) {
        float acc = 0.f;
        for (int d = 0; d < HS; d++)
          acc += (float)Qs[r][d] * fkw[(size_t)(h * HS + d) * F1 + f];
        val = (acc + fkb[(size_t)h * F1 + f]) * scale;
      }
      fq[r][lane] = val;
    }
  }
  __syncthreads();

  v8f o0 = vzero8(), o1 = vzero8(), o2 = vzero8(), o3 = vzero8();

  int cmax = t0 + 31; if (cmax > T - 1) cmax = T - 1;
  const int jbN = cmax / 32 + 1;

  for (int jb = 0; jb < jbN; jb++) {
    const int c0 = jb * 32;
    // S = (q @ K^T) * scale for 32 keys, via 2x(2 wmma); K rows read directly
    #pragma unroll
    for (int ch = 0; ch < 2; ch++) {
      const bf16* kbase =
          qkv + (size_t)(c0 + ch * 16 + lo) * C3 + Cc + h * HS + hi * 16;
      v16bf kb0 = ldfragB(kbase);
      v16bf kb1 = ldfragB(kbase + 32);
      v8f s = vzero8();
      s = wmma_bf16(qa0, kb0, s);
      s = wmma_bf16(qa1, kb1, s);
      #pragma unroll
      for (int r = 0; r < 8; r++)
        Sl[r + hi * 8][ch * 16 + lo] = s[r] * scale;
    }
    __syncthreads();

    // online softmax bookkeeping, lane r owns row r
    if (lane < 16) {
      const int r = lane, t = t0 + r;
      float m_old = mrow[r];
      float v[32];
      float bm = NEG_INF;
      #pragma unroll
      for (int cc = 0; cc < 32; cc++) {
        int c = c0 + cc;
        float xv;
        if (c <= t)            xv = Sl[r][cc];            // causal qk
        else if (c <= t + 16)  xv = fq[r][c - 1 - t0];    // future band
        else                   xv = NEG_INF;              // masked
        v[cc] = xv;
        bm = fmaxf(bm, xv);
      }
      float m_new = fmaxf(m_old, bm);
      float sc = __expf(m_old - m_new);
      float ssum = 0.f;
      #pragma unroll
      for (int cc = 0; cc < 32; cc++) {
        float e = __expf(v[cc] - m_new);
        ssum += e;
        int c = c0 + cc;
        Pl[r][cc] = (bf16)((c <= t) ? e : 0.f);  // band excluded from P@V
      }
      mrow[r] = m_new;
      lrow[r] = lrow[r] * sc + ssum;
      rsc[r]  = sc;
    }
    // stage V transposed: Vt[d][key]
    {
      const bf16* vrow = qkv + (size_t)(c0 + lane) * C3 + 2 * Cc + h * HS;
      for (int d = 0; d < HS; d++) Vt[d][lane] = vrow[d];
    }
    __syncthreads();

    // rescale O accumulators per row, then O += P @ V
    #pragma unroll
    for (int r = 0; r < 8; r++) {
      float f = rsc[r + hi * 8];
      o0[r] *= f; o1[r] *= f; o2[r] *= f; o3[r] *= f;
    }
    v16bf pf  = ldfragA(&Pl[lo][hi * 16]);
    v16bf vf0 = ldfragB(&Vt[ 0 + lo][hi * 16]);
    v16bf vf1 = ldfragB(&Vt[16 + lo][hi * 16]);
    v16bf vf2 = ldfragB(&Vt[32 + lo][hi * 16]);
    v16bf vf3 = ldfragB(&Vt[48 + lo][hi * 16]);
    o0 = wmma_bf16(pf, vf0, o0);
    o1 = wmma_bf16(pf, vf1, o1);
    o2 = wmma_bf16(pf, vf2, o2);
    o3 = wmma_bf16(pf, vf3, o3);
    __syncthreads();
  }

  // dump O, then finalize: out = O/l + sum_band exp(fq-m)/l * fv_w + fv_b
  #pragma unroll
  for (int r = 0; r < 8; r++) {
    int row = r + hi * 8;
    Ol[row][ 0 + lo] = o0[r];
    Ol[row][16 + lo] = o1[r];
    Ol[row][32 + lo] = o2[r];
    Ol[row][48 + lo] = o3[r];
  }
  __syncthreads();

  for (int idx = lane; idx < 16 * 64; idx += 32) {
    int r = idx >> 6, d = idx & 63;
    int t = t0 + r;
    float l = lrow[r], m = mrow[r];
    float acc = Ol[r][d];
    #pragma unroll
    for (int i = 0; i < 16; i++) {
      int f = t + i;
      if (f < F1)
        acc += __expf(fq[r][r + i] - m) * fvw[((size_t)h * F1 + f) * HS + d];
    }
    float val = acc / l + fvb[h * HS + d];
    out[(size_t)t * Cc + h * HS + d] = (bf16)val;
  }
}

extern "C" void kernel_launch(void* const* d_in, const int* in_sizes, int n_in,
                              void* d_out, int out_size, void* d_ws, size_t ws_size,
                              hipStream_t stream) {
  (void)in_sizes; (void)n_in; (void)out_size; (void)ws_size;
  const int T = 2048, C = 768, C3 = 2304, C4 = 3072, H = 12;

  const float* x      = (const float*)d_in[0];
  const float* ln1_w  = (const float*)d_in[1];
  const float* ln1_b  = (const float*)d_in[2];
  const float* attn_w = (const float*)d_in[3];
  const float* attn_b = (const float*)d_in[4];
  const float* fk_w   = (const float*)d_in[5];
  const float* fk_b   = (const float*)d_in[6];
  const float* fv_w   = (const float*)d_in[7];
  const float* fv_b   = (const float*)d_in[8];
  const float* proj_w = (const float*)d_in[9];
  const float* proj_b = (const float*)d_in[10];
  const float* ln2_w  = (const float*)d_in[11];
  const float* ln2_b  = (const float*)d_in[12];
  const float* ff_w1  = (const float*)d_in[13];
  const float* ff_b1  = (const float*)d_in[14];
  const float* ff_w2  = (const float*)d_in[15];
  const float* ff_b2  = (const float*)d_in[16];

  char* ws = (char*)d_ws;
  size_t off = 0;
  auto alloc = [&](size_t bytes) -> char* {
    char* p = ws + off;
    off += (bytes + 255) & ~(size_t)255;
    return p;
  };

  bf16* wT_qkv  = (bf16*)alloc((size_t)C  * C3 * 2);  // [2304][768]
  bf16* wT_proj = (bf16*)alloc((size_t)C  * C  * 2);  // [768][768]
  bf16* wT_ff1  = (bf16*)alloc((size_t)C  * C4 * 2);  // [3072][768]
  bf16* wT_ff2  = (bf16*)alloc((size_t)C4 * C  * 2);  // [768][3072]
  bf16* h_bf    = (bf16*)alloc((size_t)T * C  * 2);
  bf16* qkv_bf  = (bf16*)alloc((size_t)T * C3 * 2);
  bf16* att_bf  = (bf16*)alloc((size_t)T * C  * 2);
  float* x1     = (float*)alloc((size_t)T * C  * 4);
  bf16* h2_bf   = (bf16*)alloc((size_t)T * C  * 2);
  bf16* ff1_bf  = (bf16*)alloc((size_t)T * C4 * 2);

  auto cvtT = [&](const float* s, bf16* d, int K, int N) {
    int n = K * N;
    cvtT_kernel<<<(n + 255) / 256, 256, 0, stream>>>(s, d, K, N);
  };
  auto gemm = [&](const bf16* A, const bf16* Bt, const float* bias,
                  const float* resid, float* outF, bf16* outB,
                  int M, int N, int K, int act) {
    dim3 g(N / 64, M / 64);
    gemm_kernel<<<g, 256, 0, stream>>>(A, Bt, bias, resid, outF, outB, M, N, K, act);
  };

  // weights fp32 -> bf16, transposed to [N][K] (once; served from L2 afterwards)
  cvtT(attn_w, wT_qkv,  C,  C3);
  cvtT(proj_w, wT_proj, C,  C);
  cvtT(ff_w1,  wT_ff1,  C,  C4);
  cvtT(ff_w2,  wT_ff2,  C4, C);

  // h = LN1(x)
  ln_kernel<<<T, 256, 0, stream>>>(x, ln1_w, ln1_b, h_bf, C);
  // qkv = h @ attn_w + attn_b
  gemm(h_bf, wT_qkv, attn_b, nullptr, nullptr, qkv_bf, T, C3, C, 0);
  // attention (causal + future band), fused flash softmax
  attn_kernel<<<dim3(H, T / 16), 32, 0, stream>>>(qkv_bf, fk_w, fk_b, fv_w, fv_b,
                                                  att_bf, T);
  // x1 = x + attn @ proj_w + proj_b
  gemm(att_bf, wT_proj, proj_b, x, x1, nullptr, T, C, C, 0);
  // h2 = LN2(x1)
  ln_kernel<<<T, 256, 0, stream>>>(x1, ln2_w, ln2_b, h2_bf, C);
  // ff1 = gelu(h2 @ ff_w1 + ff_b1)
  gemm(h2_bf, wT_ff1, ff_b1, nullptr, nullptr, ff1_bf, T, C4, C, 1);
  // out = x1 + ff1 @ ff_w2 + ff_b2
  gemm(ff1_bf, wT_ff2, ff_b2, x1, (float*)d_out, nullptr, T, C, C4, 0);
}
